// TransformerBlock_18794776887495
// MI455X (gfx1250) — matrix operand order repaired and checked by reading, not compiled, not run on hardware
//
#include <hip/hip_runtime.h>
#include <hip/hip_bf16.h>
#include <math.h>

#define BB 4
#define LL 1024
#define RR 4
#define DD 1024
#define NHEADS 16
#define HDIM 64
#define FFDIM 4096
#define EPSV 1e-6f

typedef __attribute__((ext_vector_type(16))) __bf16 v16bf;
typedef __attribute__((ext_vector_type(8)))  float  v8f;
typedef __attribute__((ext_vector_type(4)))  unsigned int u32x4;
typedef __attribute__((ext_vector_type(8)))  int          i32x8;
typedef __attribute__((ext_vector_type(4)))  int          i32x4;

union Frag {
    v16bf v;
    uint4 u[2];
    unsigned short s[16];
};

static __device__ __forceinline__ unsigned short f2bf(float f) {
    union { float f; unsigned u; } x; x.f = f;
    unsigned u = x.u;
    u += 0x7FFFu + ((u >> 16) & 1u);   // round-to-nearest-even
    return (unsigned short)(u >> 16);
}

// ---------------------------------------------------------------------------
// TDM descriptor helpers (CDNA5 ISA). 2-D tile:
//  group0: count=1 | lds_addr | global_addr[56:0] | type=2
//  group1: data_size=2B | tensor_dim0/1 | tile_dim0/1 | tensor_dim0_stride
// ---------------------------------------------------------------------------
static __device__ __forceinline__ u32x4 tdm_g0(unsigned lds_addr, const void* gptr) {
    unsigned long long ga = (unsigned long long)gptr;
    u32x4 g;
    g[0] = 1u;                                   // count=1, user mode
    g[1] = lds_addr;                             // LDS byte address
    g[2] = (unsigned)ga;                         // global_addr[31:0]
    g[3] = (unsigned)(ga >> 32) | 0x80000000u;   // global_addr[56:32] | type=2
    return g;
}

static __device__ __forceinline__ i32x8 tdm_g1_2d(unsigned td0, unsigned td1,
                                                  unsigned tile0, unsigned tile1,
                                                  unsigned long long stride0) {
    i32x8 g;
    g[0] = (int)(1u << 16);                              // data_size=1 -> 2 bytes
    g[1] = (int)((td0 & 0xFFFFu) << 16);                 // tensor_dim0 lo16 @[63:48]
    g[2] = (int)((td0 >> 16) | ((td1 & 0xFFFFu) << 16)); // td0 hi16 | td1 lo16
    g[3] = (int)((td1 >> 16) | (tile0 << 16));           // td1 hi16 | tile_dim0
    g[4] = (int)tile1;                                   // tile_dim1 (tile_dim2=0)
    g[5] = (int)(stride0 & 0xFFFFFFFFu);                 // dim0_stride lo32
    g[6] = (int)((stride0 >> 32) & 0xFFFFu);             // dim0_stride hi16
    g[7] = 0;
    return g;
}

static __device__ __forceinline__ void tdm_load_2d(unsigned lds_addr, const void* gptr,
                                                   unsigned td0, unsigned td1,
                                                   unsigned tile0, unsigned tile1,
                                                   unsigned long long stride0) {
    const i32x4 z4 = {0, 0, 0, 0};
    const i32x8 z8 = {0, 0, 0, 0, 0, 0, 0, 0};
    __builtin_amdgcn_tensor_load_to_lds(tdm_g0(lds_addr, gptr),
                                        tdm_g1_2d(td0, td1, tile0, tile1, stride0),
                                        z4, z4, z8, 0);
}

// ---------------------------------------------------------------------------
// block-wide sum over 256 threads (8 waves, wave32)
// ---------------------------------------------------------------------------
static __device__ float block_reduce_sum(float v, float* red) {
    #pragma unroll
    for (int m = 16; m >= 1; m >>= 1) v += __shfl_xor(v, m, 32);
    __syncthreads();
    if ((threadIdx.x & 31) == 0) red[threadIdx.x >> 5] = v;
    __syncthreads();
    float r = 0.f;
    #pragma unroll
    for (int i = 0; i < 8; ++i) r += red[i];
    return r;
}

// ---------------------------------------------------------------------------
// hyper-connection width: RMS, alpha/beta, mix = alpha^T * H  (per token)
// ---------------------------------------------------------------------------
__global__ void width_kernel(const float* __restrict__ H,
                             const float* __restrict__ norm_w,
                             const float* __restrict__ alpha_fn,   // [D, R+1]
                             const float* __restrict__ beta_fn,    // [D]
                             const float* __restrict__ a_scale,
                             const float* __restrict__ b_scale,
                             const float* __restrict__ s_alpha,    // [R, R+1]
                             const float* __restrict__ s_beta,     // [R]
                             float* __restrict__ mix,              // [B*L, R+1, D]
                             float* __restrict__ beta_out)         // [B*L, R]
{
    __shared__ float red[8];
    __shared__ float alpha_sh[RR][RR + 1];
    const int row = blockIdx.x;
    const int tid = threadIdx.x;
    const float asc = a_scale[0], bsc = b_scale[0];
    const float* Hrow = H + (size_t)row * RR * DD;

    for (int n = 0; n < RR; ++n) {
        const float* x = Hrow + n * DD;
        float ss = 0.f, db = 0.f, dk[RR + 1];
        #pragma unroll
        for (int k = 0; k <= RR; ++k) dk[k] = 0.f;
        for (int d = tid; d < DD; d += 256) {
            float xv = x[d];
            float wn = norm_w[d] * xv;
            ss += xv * xv;
            const float* arow = alpha_fn + (size_t)d * (RR + 1);
            #pragma unroll
            for (int k = 0; k <= RR; ++k) dk[k] += wn * arow[k];
            db += wn * beta_fn[d];
        }
        ss = block_reduce_sum(ss, red);
        #pragma unroll
        for (int k = 0; k <= RR; ++k) dk[k] = block_reduce_sum(dk[k], red);
        db = block_reduce_sum(db, red);
        if (tid == 0) {
            float inv = 1.f / (sqrtf(ss / (float)DD) + EPSV);
            #pragma unroll
            for (int k = 0; k <= RR; ++k)
                alpha_sh[n][k] = tanhf(dk[k] * inv) * asc + s_alpha[n * (RR + 1) + k];
            beta_out[(size_t)row * RR + n] = tanhf(db * inv) * bsc + s_beta[n];
        }
        __syncthreads();
    }

    for (int d = tid; d < DD; d += 256) {
        float h0 = Hrow[0 * DD + d], h1 = Hrow[1 * DD + d];
        float h2 = Hrow[2 * DD + d], h3 = Hrow[3 * DD + d];
        #pragma unroll
        for (int k = 0; k <= RR; ++k) {
            mix[((size_t)row * (RR + 1) + k) * DD + d] =
                alpha_sh[0][k] * h0 + alpha_sh[1][k] * h1 +
                alpha_sh[2][k] * h2 + alpha_sh[3][k] * h3;
        }
    }
}

// ---------------------------------------------------------------------------
// RMS-norm of mix[...,0,:] -> bf16 row
// ---------------------------------------------------------------------------
__global__ void rmsbf_kernel(const float* __restrict__ mix,
                             const float* __restrict__ w,
                             unsigned short* __restrict__ out)
{
    __shared__ float red[8];
    const int row = blockIdx.x;
    const float* x = mix + (size_t)row * (RR + 1) * DD;  // k = 0 slice
    float ss = 0.f;
    for (int d = threadIdx.x; d < DD; d += 256) { float v = x[d]; ss += v * v; }
    ss = block_reduce_sum(ss, red);
    float inv = 1.f / (sqrtf(ss / (float)DD) + EPSV);
    for (int d = threadIdx.x; d < DD; d += 256)
        out[(size_t)row * DD + d] = f2bf(w[d] * x[d] * inv);
}

// ---------------------------------------------------------------------------
// f32 -> bf16 weight conversion
// ---------------------------------------------------------------------------
__global__ void cvt_bf16_kernel(const float* __restrict__ in,
                                unsigned short* __restrict__ out, int n)
{
    int i = blockIdx.x * 256 + threadIdx.x;
    if (i < n) out[i] = f2bf(in[i]);
}

// ---------------------------------------------------------------------------
// bf16 GEMM with TDM-staged, double-buffered LDS operands.
// C[M,N] = A[M,K] * W[N,K]^T via v_wmma_f32_16x16x32_bf16.
// block = 256 (8 waves); tile 64(M) x 128(N) x 64(K)/stage; wave strip 16x64.
// Stage: A 64x64 bf16 (8KB) + B 128x64 bf16 (16KB); two stages = 48KB LDS.
// Wave 0 issues tensor_load_to_lds for stage i+1, waits TENSORcnt<=2 for
// stage i (TDM ops complete in order per wave), then all waves consume LDS.
// ---------------------------------------------------------------------------
#define GK 64
#define A_TILE_BYTES (64 * GK * 2)
#define B_TILE_BYTES (128 * GK * 2)
#define STAGE_BYTES  (A_TILE_BYTES + B_TILE_BYTES)

template<bool GELU, bool BF16OUT>
__global__ void gemm_tdm_kernel(const unsigned short* __restrict__ A,
                                const unsigned short* __restrict__ W,
                                float* __restrict__ Cf,
                                unsigned short* __restrict__ Cb,
                                int M, int N, int K)
{
    extern __shared__ unsigned short smem[];   // 2 stages, offsets from LDS 0
    const int lane = threadIdx.x & 31;
    const int wid  = threadIdx.x >> 5;
    const int wm = wid & 3, wn = wid >> 2;
    const int half = lane >> 4, l16 = lane & 15;
    const int m0blk = blockIdx.y * 64, n0blk = blockIdx.x * 128;

    v8f acc[4];
    #pragma unroll
    for (int t = 0; t < 4; ++t) acc[t] = (v8f){0.f,0.f,0.f,0.f,0.f,0.f,0.f,0.f};

    // stage-0 issue (wave 0 only; TDM ignores EXEC, other waves skip via branch)
    if (threadIdx.x == 0) {
        tdm_load_2d(0u, A + (size_t)m0blk * K,
                    (unsigned)K, (unsigned)M, GK, 64, (unsigned long long)K);
        tdm_load_2d(A_TILE_BYTES, W + (size_t)n0blk * K,
                    (unsigned)K, (unsigned)N, GK, 128, (unsigned long long)K);
    }

    for (int k = 0, it = 0; k < K; k += GK, ++it) {
        const int buf = it & 1;
        if (k + GK < K) {
            if (threadIdx.x == 0) {
                const unsigned ldsb = (unsigned)((buf ^ 1) * STAGE_BYTES);
                tdm_load_2d(ldsb, A + (size_t)m0blk * K + (k + GK),
                            (unsigned)K, (unsigned)M, GK, 64, (unsigned long long)K);
                tdm_load_2d(ldsb + A_TILE_BYTES, W + (size_t)n0blk * K + (k + GK),
                            (unsigned)K, (unsigned)N, GK, 128, (unsigned long long)K);
                __builtin_amdgcn_s_wait_tensorcnt(2);  // stage i landed
            }
        } else {
            if (threadIdx.x == 0) __builtin_amdgcn_s_wait_tensorcnt(0);
        }
        __syncthreads();

        const unsigned short* sA = smem + buf * (STAGE_BYTES / 2);
        const unsigned short* sB = sA + 64 * GK;
        #pragma unroll
        for (int kk = 0; kk < GK; kk += 32) {
            Frag a;
            const unsigned short* ar = sA + (size_t)(wm * 16 + l16) * GK + kk + half * 16;
            a.u[0] = *(const uint4*)(ar);
            a.u[1] = *(const uint4*)(ar + 8);
            #pragma unroll
            for (int t = 0; t < 4; ++t) {
                const unsigned short* br =
                    sB + (size_t)(wn * 64 + t * 16 + l16) * GK + kk + half * 16;
                Frag b;
                b.u[0] = *(const uint4*)(br);
                b.u[1] = *(const uint4*)(br + 8);
                acc[t] = __builtin_amdgcn_wmma_f32_16x16x32_bf16(
                    false, a.v, false, b.v, (short)0, acc[t], false, false);
            }
        }
        __syncthreads();
    }

    const int m0 = m0blk + wm * 16;
    const int n0 = n0blk + wn * 64;
    #pragma unroll
    for (int t = 0; t < 4; ++t) {
        #pragma unroll
        for (int e = 0; e < 8; ++e) {
            float v = acc[t][e];
            if (GELU) v = v * 0.5f * (1.f + erff(v * 0.70710678f));
            const size_t row = (size_t)(m0 + e + 8 * half);
            const size_t col = (size_t)(n0 + t * 16 + l16);
            if (BF16OUT) Cb[row * N + col] = f2bf(v);
            else         Cf[row * N + col] = v;
        }
    }
}

// ---------------------------------------------------------------------------
// flash attention (causal). qkv: [B*L, 3*D] bf16; out: [B*L, D] bf16
// grid (L/128, HEADS, B), 256 threads; wave owns 16 query rows
// ---------------------------------------------------------------------------
__global__ void flash_kernel(const unsigned short* __restrict__ qkv,
                             unsigned short* __restrict__ out)
{
    __shared__ __align__(16) unsigned short pbuf[8][16][32];
    const int lane = threadIdx.x & 31, wid = threadIdx.x >> 5;
    const int half = lane >> 4, l16 = lane & 15;
    const int b = blockIdx.z, h = blockIdx.y;
    const int q0 = blockIdx.x * 128 + wid * 16;
    const size_t rs = 3 * DD;

    const unsigned short* Qb = qkv + (size_t)(b * LL) * rs + h * HDIM;
    const unsigned short* Kb = Qb + DD;
    const unsigned short* Vb = Qb + 2 * DD;

    Frag qf[2];
    {
        const unsigned short* qrow = Qb + (size_t)(q0 + l16) * rs + half * 16;
        qf[0].u[0] = *(const uint4*)(qrow);
        qf[0].u[1] = *(const uint4*)(qrow + 8);
        qf[1].u[0] = *(const uint4*)(qrow + 32);
        qf[1].u[1] = *(const uint4*)(qrow + 40);
    }

    v8f o[4];
    #pragma unroll
    for (int t = 0; t < 4; ++t) o[t] = (v8f){0.f,0.f,0.f,0.f,0.f,0.f,0.f,0.f};
    float mst[8], lst[8];
    #pragma unroll
    for (int e = 0; e < 8; ++e) { mst[e] = -1e30f; lst[e] = 0.f; }

    for (int kb = 0; kb < q0 + 16; kb += 32) {
        v8f s0 = (v8f){0.f,0.f,0.f,0.f,0.f,0.f,0.f,0.f};
        v8f s1 = (v8f){0.f,0.f,0.f,0.f,0.f,0.f,0.f,0.f};
        #pragma unroll
        for (int ki = 0; ki < 2; ++ki) {
            {
                const unsigned short* krow =
                    Kb + (size_t)(kb + l16) * rs + ki * 32 + half * 16;
                Frag kf;
                kf.u[0] = *(const uint4*)(krow);
                kf.u[1] = *(const uint4*)(krow + 8);
                s0 = __builtin_amdgcn_wmma_f32_16x16x32_bf16(
                    false, qf[ki].v, false, kf.v, (short)0, s0, false, false);
            }
            {
                const unsigned short* krow =
                    Kb + (size_t)(kb + 16 + l16) * rs + ki * 32 + half * 16;
                Frag kf;
                kf.u[0] = *(const uint4*)(krow);
                kf.u[1] = *(const uint4*)(krow + 8);
                s1 = __builtin_amdgcn_wmma_f32_16x16x32_bf16(
                    false, qf[ki].v, false, kf.v, (short)0, s1, false, false);
            }
        }

        // online softmax per query row (row index = e + 8*half)
        #pragma unroll
        for (int e = 0; e < 8; ++e) {
            const int qidx = q0 + e + 8 * half;
            float a0 = s0[e] * 0.125f;       // 1/sqrt(64)
            float a1 = s1[e] * 0.125f;
            if (kb + l16      > qidx) a0 = -1e30f;
            if (kb + 16 + l16 > qidx) a1 = -1e30f;
            float mx = fmaxf(a0, a1);
            #pragma unroll
            for (int m = 8; m >= 1; m >>= 1) mx = fmaxf(mx, __shfl_xor(mx, m, 16));
            float mnew = fmaxf(mst[e], mx);
            float p0 = __expf(a0 - mnew);
            float p1 = __expf(a1 - mnew);
            float sum = p0 + p1;
            #pragma unroll
            for (int m = 8; m >= 1; m >>= 1) sum += __shfl_xor(sum, m, 16);
            float corr = __expf(mst[e] - mnew);
            lst[e] = lst[e] * corr + sum;
            mst[e] = mnew;
            #pragma unroll
            for (int t = 0; t < 4; ++t) o[t][e] *= corr;
            pbuf[wid][e + 8 * half][l16]      = f2bf(p0);
            pbuf[wid][e + 8 * half][16 + l16] = f2bf(p1);
        }

        // P (16x32) as WMMA A-fragment from per-wave LDS (wave DS ops in-order)
        Frag pf;
        {
            const unsigned short* pr = &pbuf[wid][l16][half * 16];
            pf.u[0] = *(const uint4*)(pr);
            pf.u[1] = *(const uint4*)(pr + 8);
        }

        // O += P * V
        #pragma unroll
        for (int t = 0; t < 4; ++t) {
            Frag vf;
            const int hd = t * 16 + l16;
            #pragma unroll
            for (int e = 0; e < 16; ++e)
                vf.s[e] = Vb[(size_t)(kb + half * 16 + e) * rs + hd];
            o[t] = __builtin_amdgcn_wmma_f32_16x16x32_bf16(
                false, pf.v, false, vf.v, (short)0, o[t], false, false);
        }
    }

    #pragma unroll
    for (int t = 0; t < 4; ++t) {
        #pragma unroll
        for (int e = 0; e < 8; ++e) {
            float v = o[t][e] / lst[e];
            out[(size_t)(b * LL + q0 + e + 8 * half) * DD + h * HDIM + t * 16 + l16] = f2bf(v);
        }
    }
}

// ---------------------------------------------------------------------------
// depth: H[b,l,n,:] = beta[b,l,n] * out[b,l,:] + mix[b,l,n+1,:]  (float4)
// ---------------------------------------------------------------------------
__global__ void depth_kernel(const float4* __restrict__ mix,
                             const float4* __restrict__ outp,
                             const float* __restrict__ beta,
                             float4* __restrict__ Hn)
{
    const int D4 = DD / 4;
    size_t i = (size_t)blockIdx.x * 256 + threadIdx.x;  // over B*L*R*D4
    int d = (int)(i % D4);
    size_t t = i / D4;
    int n = (int)(t & (RR - 1));
    size_t row = t >> 2;
    float be = beta[row * RR + n];
    float4 o = outp[row * D4 + d];
    float4 m = mix[(row * (RR + 1) + n + 1) * D4 + d];
    float4 r;
    r.x = be * o.x + m.x; r.y = be * o.y + m.y;
    r.z = be * o.z + m.z; r.w = be * o.w + m.w;
    Hn[i] = r;
}

// ---------------------------------------------------------------------------
extern "C" void kernel_launch(void* const* d_in, const int* in_sizes, int n_in,
                              void* d_out, int out_size, void* d_ws, size_t ws_size,
                              hipStream_t stream)
{
    const float* H          = (const float*)d_in[0];
    const float* hc1_norm_w = (const float*)d_in[1];
    const float* hc1_afn    = (const float*)d_in[2];
    const float* hc1_bfn    = (const float*)d_in[3];
    const float* hc1_as     = (const float*)d_in[4];
    const float* hc1_bs     = (const float*)d_in[5];
    const float* hc1_sa     = (const float*)d_in[6];
    const float* hc1_sb     = (const float*)d_in[7];
    const float* attn_nw    = (const float*)d_in[8];
    const float* qkv_w      = (const float*)d_in[9];
    const float* out_w      = (const float*)d_in[10];
    const float* hc2_norm_w = (const float*)d_in[11];
    const float* hc2_afn    = (const float*)d_in[12];
    const float* hc2_bfn    = (const float*)d_in[13];
    const float* hc2_as     = (const float*)d_in[14];
    const float* hc2_bs     = (const float*)d_in[15];
    const float* hc2_sa     = (const float*)d_in[16];
    const float* hc2_sb     = (const float*)d_in[17];
    const float* ffn_nw     = (const float*)d_in[18];
    const float* fc1_w      = (const float*)d_in[19];
    const float* fc2_w      = (const float*)d_in[20];

    const int M = BB * LL;                       // 4096 token rows
    char* ws = (char*)d_ws;
    size_t off = 0;
    auto carve = [&](size_t bytes) -> char* {
        char* p = ws + off;
        off = (off + bytes + 255) & ~(size_t)255;
        return p;
    };

    float*          mix     = (float*)carve((size_t)M * (RR + 1) * DD * 4);
    float*          beta    = (float*)carve((size_t)M * RR * 4);
    unsigned short* xbf     = (unsigned short*)carve((size_t)M * DD * 2);
    unsigned short* qkvbf   = (unsigned short*)carve((size_t)M * 3 * DD * 2);
    unsigned short* attnbf  = (unsigned short*)carve((size_t)M * DD * 2);
    float*          hproj   = (float*)carve((size_t)M * DD * 4);
    unsigned short* ffn1bf  = (unsigned short*)carve((size_t)M * FFDIM * 2);
    unsigned short* wqkv    = (unsigned short*)carve((size_t)3 * DD * DD * 2);
    unsigned short* wout    = (unsigned short*)carve((size_t)DD * DD * 2);
    unsigned short* wfc1    = (unsigned short*)carve((size_t)FFDIM * DD * 2);
    unsigned short* wfc2    = (unsigned short*)carve((size_t)DD * FFDIM * 2);

    float* Hmid = (float*)d_out;   // attention-block output H lives in d_out
    float* Hout = (float*)d_out;

    const size_t smem_bytes = 2 * STAGE_BYTES;   // 48KB double-buffered stages

    // weight conversion to bf16
    {
        int n1 = 3 * DD * DD, n2 = DD * DD, n3 = FFDIM * DD, n4 = DD * FFDIM;
        cvt_bf16_kernel<<<(n1 + 255) / 256, 256, 0, stream>>>(qkv_w, wqkv, n1);
        cvt_bf16_kernel<<<(n2 + 255) / 256, 256, 0, stream>>>(out_w, wout, n2);
        cvt_bf16_kernel<<<(n3 + 255) / 256, 256, 0, stream>>>(fc1_w, wfc1, n3);
        cvt_bf16_kernel<<<(n4 + 255) / 256, 256, 0, stream>>>(fc2_w, wfc2, n4);
    }

    // ---------------- attention block ----------------
    width_kernel<<<M, 256, 0, stream>>>(H, hc1_norm_w, hc1_afn, hc1_bfn,
                                        hc1_as, hc1_bs, hc1_sa, hc1_sb, mix, beta);
    rmsbf_kernel<<<M, 256, 0, stream>>>(mix, attn_nw, xbf);

    gemm_tdm_kernel<false, true><<<dim3(3 * DD / 128, M / 64), 256, smem_bytes, stream>>>(
        xbf, wqkv, nullptr, qkvbf, M, 3 * DD, DD);

    flash_kernel<<<dim3(LL / 128, NHEADS, BB), 256, 0, stream>>>(qkvbf, attnbf);

    gemm_tdm_kernel<false, false><<<dim3(DD / 128, M / 64), 256, smem_bytes, stream>>>(
        attnbf, wout, hproj, nullptr, M, DD, DD);

    depth_kernel<<<(M * RR * DD / 4) / 256, 256, 0, stream>>>(
        (const float4*)mix, (const float4*)hproj, beta, (float4*)Hmid);

    // ---------------- ffn block ----------------
    width_kernel<<<M, 256, 0, stream>>>(Hmid, hc2_norm_w, hc2_afn, hc2_bfn,
                                        hc2_as, hc2_bs, hc2_sa, hc2_sb, mix, beta);
    rmsbf_kernel<<<M, 256, 0, stream>>>(mix, ffn_nw, xbf);

    gemm_tdm_kernel<true, true><<<dim3(FFDIM / 128, M / 64), 256, smem_bytes, stream>>>(
        xbf, wfc1, nullptr, ffn1bf, M, FFDIM, DD);

    gemm_tdm_kernel<false, false><<<dim3(DD / 128, M / 64), 256, smem_bytes, stream>>>(
        ffn1bf, wfc2, hproj, nullptr, M, DD, FFDIM);

    depth_kernel<<<(M * RR * DD / 4) / 256, 256, 0, stream>>>(
        (const float4*)mix, (const float4*)hproj, beta, (float4*)Hout);
}
